// DCRNNModel_87909390614974
// MI455X (gfx1250) — matrix operand order repaired and checked by reading, not compiled, not run on hardware
//
#include <hip/hip_runtime.h>
#include <math.h>

typedef __attribute__((ext_vector_type(16))) _Float16 v16h;
typedef __attribute__((ext_vector_type(8)))  float    v8f;

#define NN   5000          // nodes
#define NE   50000         // edges
#define BB   8             // batch
#define TT   12            // seq
#define INC  2             // input channels
#define HH   64            // hidden
#define KCH  3             // Chebyshev K
#define MR   (NN*BB)       // 40000 rows
#define NTIL (MR/16)       // 2500 row tiles of 16

static __device__ __forceinline__ float sigmoidf_(float x) {
  return 1.0f / (1.0f + __expf(-x));
}

// ---------------- graph normalization (once per launch) ----------------
__global__ void k_degree(const int* __restrict__ src, const int* __restrict__ dst,
                         float* __restrict__ degO, float* __restrict__ degI) {
  int e = blockIdx.x * blockDim.x + threadIdx.x;
  if (e >= NE) return;
  atomicAdd(&degO[src[e]], 1.0f);
  atomicAdd(&degI[dst[e]], 1.0f);
}

__global__ void k_norm(const int* __restrict__ src, const int* __restrict__ dst,
                       const float* __restrict__ degO, const float* __restrict__ degI,
                       float* __restrict__ nO, float* __restrict__ nI) {
  int e = blockIdx.x * blockDim.x + threadIdx.x;
  if (e >= NE) return;
  float d0 = degO[src[e]];
  float d1 = degI[dst[e]];
  nO[e] = d0 > 0.0f ? 1.0f / d0 : 0.0f;
  nI[e] = d1 > 0.0f ? 1.0f / d1 : 0.0f;
}

// ---------------- weight pack: [2,K,Fin,64] f32 -> WMMA B-fragment f16 ----
// Concatenated K-dim: seg0 = W[0,0]+W[1,0], seg1 = W[0,1], seg2 = W[1,1],
// seg3 = W[0,2], seg4 = W[1,2]; padded to nsteps*32. Layout:
// Wp[cg][s][lane][e] where B(K,col): col = cg*16+(lane&15), K = s*32 + (lane<16?0:8) + (e<8 ? e : 16+(e-8))
__global__ void k_pack_w(const float* __restrict__ W, _Float16* __restrict__ Wp,
                         int Fin, int nsteps) {
  int i = blockIdx.x * blockDim.x + threadIdx.x;
  int total = 4 * nsteps * 32 * 16;
  if (i >= total) return;
  int e    = i & 15;
  int lane = (i >> 4) & 31;
  int s    = (i >> 9) % nsteps;
  int cg   = (i >> 9) / nsteps;
  int K    = s * 32 + (lane < 16 ? 0 : 8) + (e < 8 ? e : 16 + (e - 8));
  int col  = cg * 16 + (lane & 15);
  float v = 0.0f;
  if (K < 5 * Fin) {
    int seg = K / Fin;
    int f   = K - seg * Fin;
    int dir = 0, kk = 0;
    if      (seg == 1) { dir = 0; kk = 1; }
    else if (seg == 2) { dir = 1; kk = 1; }
    else if (seg == 3) { dir = 0; kk = 2; }
    else if (seg == 4) { dir = 1; kk = 2; }
    v = W[(((size_t)dir * KCH + kk) * Fin + f) * HH + col];
    if (seg == 0) v += W[(((size_t)1 * KCH + 0) * Fin + f) * HH + col];
  }
  Wp[i] = (_Float16)v;
}

// ---------------- build concatenated input [MR, Fin] ----------------
// f < split : layer input (x[b,t,n,:] for layer0, H0 for layer1)
// f >= split: hidden state (optionally gated by R)
__global__ void k_build_xh(const float* __restrict__ xg, int t,
                           const float* __restrict__ Hin,
                           const float* __restrict__ Hst,
                           const float* __restrict__ Rg,
                           float* __restrict__ XH, int Fin, int split, int use_x) {
  int i = blockIdx.x * blockDim.x + threadIdx.x;
  if (i >= MR * Fin) return;
  int row = i / Fin, f = i - row * Fin;
  float v;
  if (f < split) {
    if (use_x) {
      int n = row / BB, b = row - n * BB;
      v = xg[((size_t)(b * TT + t) * NN + n) * INC + f];
    } else {
      v = Hin[(size_t)row * HH + f];
    }
  } else {
    int fh = f - split;
    v = Hst[(size_t)row * HH + fh];
    if (Rg) v *= Rg[(size_t)row * HH + fh];
  }
  XH[i] = v;
}

// ---------------- sparse diffusion: out[dst] += norm[e] * in[src] ----------
// Rows are node-major: node block = BB*Fin contiguous floats. Coalesced j.
__global__ void k_prop(const float* __restrict__ in, float* __restrict__ out,
                       const int* __restrict__ src, const int* __restrict__ dst,
                       const float* __restrict__ nrm, int BF) {
  int e = blockIdx.y;
  int j = blockIdx.x * blockDim.x + threadIdx.x;
  if (j >= BF) return;
  float nv = nrm[e];
  if (nv == 0.0f) return;
  atomicAdd(&out[(size_t)dst[e] * BF + j], nv * in[(size_t)src[e] * BF + j]);
}

// ---------------- pack A operand into WMMA 16x32 f16 fragments ----------
// Ap dword i: d=i&7, lane=(i>>3)&31, s=(i>>8)%nsteps, tile=(i>>8)/nsteps
// row = tile*16 + (lane&15); kb = lane<16?0:8; K pair per ISA A-layout.
// K segments: 0=XH, 1=Tx1o, 2=Tx1i, 3=2*P2o-XH, 4=2*P2i-XH (Chebyshev T2).
__global__ void k_pack_a(const float* __restrict__ XH,
                         const float* __restrict__ P1o, const float* __restrict__ P1i,
                         const float* __restrict__ P2o, const float* __restrict__ P2i,
                         unsigned int* __restrict__ Ap, int Fin, int nsteps) {
  int i = blockIdx.x * blockDim.x + threadIdx.x;
  int total = NTIL * nsteps * 256;
  if (i >= total) return;
  int d    = i & 7;
  int lane = (i >> 3) & 31;
  int tmp  = i >> 8;
  int s    = tmp % nsteps;
  int tile = tmp / nsteps;
  int row  = tile * 16 + (lane & 15);
  int kb   = (lane < 16) ? 0 : 8;
  int e2   = 2 * d;
  int K0   = s * 32 + kb + (e2 < 8 ? e2 : 16 + (e2 - 8));
  float v[2];
#pragma unroll
  for (int q = 0; q < 2; ++q) {
    int K = K0 + q;
    float r = 0.0f;
    if (K < 5 * Fin) {
      int seg = K / Fin, f = K - seg * Fin;
      size_t o = (size_t)row * Fin + f;
      if      (seg == 0) r = XH[o];
      else if (seg == 1) r = P1o[o];
      else if (seg == 2) r = P1i[o];
      else if (seg == 3) r = 2.0f * P2o[o] - XH[o];
      else               r = 2.0f * P2i[o] - XH[o];
    }
    v[q] = r;
  }
  union { _Float16 h[2]; unsigned int u; } pk;
  pk.h[0] = (_Float16)v[0];
  pk.h[1] = (_Float16)v[1];
  Ap[i] = pk.u;
}

// ---------------- fused Z/R gate GEMM: sigmoid(A@Wz+bz), sigmoid(A@Wr+br) ---
__global__ void __launch_bounds__(256) k_gemm_gates(
    const v16h* __restrict__ Ap, const v16h* __restrict__ Wz, const v16h* __restrict__ Wr,
    const float* __restrict__ bz, const float* __restrict__ br,
    float* __restrict__ Z, float* __restrict__ R, int nsteps) {
  int wave = threadIdx.x >> 5, lane = threadIdx.x & 31;
  int tile = blockIdx.x * 8 + wave;
  if (tile >= NTIL) return;                 // wave-uniform: EXEC stays all-ones for WMMA
  v8f cz[4] = {}, cr[4] = {};
  const v16h* a = Ap + (size_t)tile * nsteps * 32 + lane;
  for (int s = 0; s < nsteps; ++s) {
    v16h av = a[(size_t)s * 32];
#pragma unroll
    for (int cg = 0; cg < 4; ++cg) {
      v16h bzf = Wz[(size_t)(cg * nsteps + s) * 32 + lane];
      v16h brf = Wr[(size_t)(cg * nsteps + s) * 32 + lane];
      cz[cg] = __builtin_amdgcn_wmma_f32_16x16x32_f16(false, av, false, bzf,
                                                      (short)0, cz[cg], false, false);
      cr[cg] = __builtin_amdgcn_wmma_f32_16x16x32_f16(false, av, false, brf,
                                                      (short)0, cr[cg], false, false);
    }
  }
  int colb = lane & 15;
  int roff = (lane < 16) ? 0 : 8;
#pragma unroll
  for (int cg = 0; cg < 4; ++cg) {
    int col = cg * 16 + colb;
    float bzv = bz[col], brv = br[col];
#pragma unroll
    for (int j = 0; j < 8; ++j) {
      size_t idx = (size_t)(tile * 16 + roff + j) * HH + col;
      Z[idx] = sigmoidf_(cz[cg][j] + bzv);
      R[idx] = sigmoidf_(cr[cg][j] + brv);
    }
  }
}

// ---------------- candidate GEMM with fused GRU update --------------------
// H = Z*H + (1-Z)*tanh(A@Wh + bh)   (in-place on H; same element per lane)
__global__ void __launch_bounds__(256) k_gemm_cand(
    const v16h* __restrict__ Ap, const v16h* __restrict__ Wh,
    const float* __restrict__ bh, const float* __restrict__ Zb,
    float* __restrict__ Hst, int nsteps) {
  int wave = threadIdx.x >> 5, lane = threadIdx.x & 31;
  int tile = blockIdx.x * 8 + wave;
  if (tile >= NTIL) return;
  v8f c[4] = {};
  const v16h* a = Ap + (size_t)tile * nsteps * 32 + lane;
  for (int s = 0; s < nsteps; ++s) {
    v16h av = a[(size_t)s * 32];
#pragma unroll
    for (int cg = 0; cg < 4; ++cg) {
      v16h bf = Wh[(size_t)(cg * nsteps + s) * 32 + lane];
      c[cg] = __builtin_amdgcn_wmma_f32_16x16x32_f16(false, av, false, bf,
                                                     (short)0, c[cg], false, false);
    }
  }
  int colb = lane & 15;
  int roff = (lane < 16) ? 0 : 8;
#pragma unroll
  for (int cg = 0; cg < 4; ++cg) {
    int col = cg * 16 + colb;
    float bv = bh[col];
#pragma unroll
    for (int j = 0; j < 8; ++j) {
      size_t idx = (size_t)(tile * 16 + roff + j) * HH + col;
      float ht = tanhf(c[cg][j] + bv);
      float z  = Zb[idx];
      float h0 = Hst[idx];
      Hst[idx] = z * h0 + (1.0f - z) * ht;
    }
  }
}

// ---------------- output projection: out[b,t,n] = H1[row]·Wo + bo ---------
__global__ void k_proj(const float* __restrict__ H1, const float* __restrict__ Wo,
                       const float* __restrict__ bo, float* __restrict__ out, int t) {
  int row = blockIdx.x * blockDim.x + threadIdx.x;
  if (row >= MR) return;
  float s = bo[0];
#pragma unroll 8
  for (int k = 0; k < HH; ++k) s += H1[(size_t)row * HH + k] * Wo[k];
  int n = row / BB, b = row - n * BB;
  out[(size_t)(b * TT + t) * NN + n] = s;
}

// ===========================================================================
extern "C" void kernel_launch(void* const* d_in, const int* in_sizes, int n_in,
                              void* d_out, int out_size, void* d_ws, size_t ws_size,
                              hipStream_t stream) {
  (void)in_sizes; (void)n_in; (void)out_size; (void)ws_size;

  const float* x   = (const float*)d_in[0];
  const int*   ei  = (const int*)d_in[1];
  const int*   src = ei;
  const int*   dst = ei + NE;
  const float* Wg[2][3]; const float* bg[2][3];
  int ii = 2;
  for (int l = 0; l < 2; ++l)
    for (int g = 0; g < 3; ++g) { Wg[l][g] = (const float*)d_in[ii++]; bg[l][g] = (const float*)d_in[ii++]; }
  const float* Wo = (const float*)d_in[ii++];
  const float* bo = (const float*)d_in[ii++];

  // ---- workspace carve (bump allocator, 256B aligned) ----
  char* p = (char*)d_ws;
  auto alloc = [&](size_t bytes) -> char* {
    char* r = p; p += (bytes + 255) & ~(size_t)255; return r;
  };
  float* H0   = (float*)alloc((size_t)MR * HH * 4);
  float* H1   = (float*)alloc((size_t)MR * HH * 4);
  float* Z    = (float*)alloc((size_t)MR * HH * 4);
  float* R    = (float*)alloc((size_t)MR * HH * 4);
  float* XH   = (float*)alloc((size_t)MR * 128 * 4);
  float* P1o  = (float*)alloc((size_t)MR * 128 * 4);
  float* P1i  = (float*)alloc((size_t)MR * 128 * 4);
  float* P2o  = (float*)alloc((size_t)MR * 128 * 4);
  float* P2i  = (float*)alloc((size_t)MR * 128 * 4);
  unsigned int* Ap = (unsigned int*)alloc((size_t)NTIL * 20 * 256 * 4);
  _Float16* Wp[2][3];
  const int FinL[2]    = { INC + HH, HH + HH };     // 66, 128
  const int nstepsL[2] = { (5*(INC+HH)+31)/32, (5*(HH+HH)+31)/32 }; // 11, 20
  for (int l = 0; l < 2; ++l)
    for (int g = 0; g < 3; ++g)
      Wp[l][g] = (_Float16*)alloc((size_t)4 * nstepsL[l] * 32 * 16 * 2);
  float* degO = (float*)alloc(NN * 4);
  float* degI = (float*)alloc(NN * 4);
  float* nO   = (float*)alloc(NE * 4);
  float* nI   = (float*)alloc(NE * 4);

  // ---- one-time per-launch init ----
  hipMemsetAsync(H0, 0, (size_t)MR * HH * 4, stream);
  hipMemsetAsync(H1, 0, (size_t)MR * HH * 4, stream);
  hipMemsetAsync(degO, 0, NN * 4, stream);
  hipMemsetAsync(degI, 0, NN * 4, stream);
  k_degree<<<(NE + 255) / 256, 256, 0, stream>>>(src, dst, degO, degI);
  k_norm<<<(NE + 255) / 256, 256, 0, stream>>>(src, dst, degO, degI, nO, nI);
  for (int l = 0; l < 2; ++l)
    for (int g = 0; g < 3; ++g) {
      int tot = 4 * nstepsL[l] * 32 * 16;
      k_pack_w<<<(tot + 255) / 256, 256, 0, stream>>>(Wg[l][g], Wp[l][g], FinL[l], nstepsL[l]);
    }

  auto run_props = [&](const float* Xbuf, int Fin) {
    int BF = BB * Fin;
    size_t bytes = (size_t)MR * Fin * 4;
    dim3 pg((BF + 255) / 256, NE);
    hipMemsetAsync(P1o, 0, bytes, stream);
    hipMemsetAsync(P1i, 0, bytes, stream);
    k_prop<<<pg, 256, 0, stream>>>(Xbuf, P1o, src, dst, nO, BF);
    k_prop<<<pg, 256, 0, stream>>>(Xbuf, P1i, src, dst, nI, BF);
    hipMemsetAsync(P2o, 0, bytes, stream);
    hipMemsetAsync(P2i, 0, bytes, stream);
    k_prop<<<pg, 256, 0, stream>>>(P1o, P2o, src, dst, nO, BF);
    k_prop<<<pg, 256, 0, stream>>>(P1i, P2i, src, dst, nI, BF);
  };

  // ---- recurrent loop ----
  for (int t = 0; t < TT; ++t) {
    for (int l = 0; l < 2; ++l) {
      int Fin = FinL[l], nsteps = nstepsL[l];
      int split = (l == 0) ? INC : HH;
      int use_x = (l == 0) ? 1 : 0;
      float* Hst = (l == 0) ? H0 : H1;
      const float* Hin = H0;   // for l==1, H0 already holds the new layer-0 state
      int nel = MR * Fin;
      int ta  = NTIL * nsteps * 256;

      // gates: XH = [Xin, H]
      k_build_xh<<<(nel + 255) / 256, 256, 0, stream>>>(x, t, Hin, Hst, nullptr, XH, Fin, split, use_x);
      run_props(XH, Fin);
      k_pack_a<<<(ta + 255) / 256, 256, 0, stream>>>(XH, P1o, P1i, P2o, P2i, Ap, Fin, nsteps);
      k_gemm_gates<<<(NTIL + 7) / 8, 256, 0, stream>>>(
          (const v16h*)Ap, (const v16h*)Wp[l][0], (const v16h*)Wp[l][1],
          bg[l][0], bg[l][1], Z, R, nsteps);

      // candidate: XH = [Xin, H*R], then fused GRU update into H
      k_build_xh<<<(nel + 255) / 256, 256, 0, stream>>>(x, t, Hin, Hst, R, XH, Fin, split, use_x);
      run_props(XH, Fin);
      k_pack_a<<<(ta + 255) / 256, 256, 0, stream>>>(XH, P1o, P1i, P2o, P2i, Ap, Fin, nsteps);
      k_gemm_cand<<<(NTIL + 7) / 8, 256, 0, stream>>>(
          (const v16h*)Ap, (const v16h*)Wp[l][2], bg[l][2], Z, Hst, nsteps);
    }
    k_proj<<<(MR + 255) / 256, 256, 0, stream>>>(H1, Wo, bo, (float*)d_out, t);
  }
}
